// HashGridMLP_76192719832104
// MI455X (gfx1250) — compile-verified, matrix operand order checked by reading, not run on hardware
//
#include <hip/hip_runtime.h>
#include <hip/hip_bf16.h>

typedef __attribute__((ext_vector_type(16))) _Float16 v16h;
typedef __attribute__((ext_vector_type(8)))  _Float16 v8h;
typedef __attribute__((ext_vector_type(8)))  float    v8f;

#define NLEVELS   16
#define TSIZE     (1u << 19)     // hash table entries per level
#define NPB       256            // points per block (8 wave32)
#define NWAVES    8

#define P2 2654435761u
#define P3 805459861u

__global__ __launch_bounds__(256)
void HashGridMLP_76192719832104_kernel(const float* __restrict__ x,
                                       const float* __restrict__ tables,
                                       const float* __restrict__ W1,
                                       const float* __restrict__ b1,
                                       const float* __restrict__ W2,
                                       const float* __restrict__ b2,
                                       float* __restrict__ out, int npts)
{
    // W1 pre-swizzled into per-lane WMMA B-fragment order: [nt][lane][16 halves]
    __shared__ __align__(16) _Float16 sW1frag[4][32][16];   // 4 KB
    __shared__ float    sB1[64];
    __shared__ float    sW2[64];
    __shared__ __align__(16) _Float16 sEnc[NWAVES][32 * 32]; // 16 KB, [row][k]

    const int tid  = threadIdx.x;
    const int wave = tid >> 5;
    const int lane = tid & 31;

    // ---- stage W1 directly in B-fragment layout (f32 -> f16) ----
    {
        const int frag    = tid >> 1;          // 0..127 = nt*32 + lane
        const int halfsel = (tid & 1) * 8;     // first or second 8 halves
        const int fnt     = frag >> 5;
        const int flane   = frag & 31;
        const int fkb     = (flane < 16) ? 0 : 8;
        const int fn      = fnt * 16 + (flane & 15);
        #pragma unroll
        for (int h = 0; h < 8; ++h) {
            const int slot = halfsel + h;
            // slots 0..7  -> K = kb + slot ; slots 8..15 -> K = kb + 16 + (slot-8)
            const int k = (slot < 8) ? (fkb + slot) : (fkb + 8 + slot);
            sW1frag[fnt][flane][slot] = (_Float16)W1[k * 64 + fn];
        }
    }
    if (tid < 64) { sB1[tid] = b1[tid]; sW2[tid] = W2[tid]; }

    // ---- hash-grid encoding: one point per lane ----
    const int gp = blockIdx.x * NPB + tid;
    float px = 0.f, py = 0.f, pz = 0.f;
    if (gp < npts) {
        px = (x[gp * 3 + 0] + 1.0f) * 0.5f;
        py = (x[gp * 3 + 1] + 1.0f) * 0.5f;
        pz = (x[gp * 3 + 2] + 1.0f) * 0.5f;
    }
    const float kRes[NLEVELS] = {16.f, 24.f, 36.f, 54.f, 81.f, 121.f, 182.f, 273.f,
                                 410.f, 615.f, 922.f, 1383.f, 2075.f, 3113.f, 4670.f, 7006.f};

    unsigned encbits[NLEVELS];   // one packed half2 per level
    #pragma unroll
    for (int l = 0; l < NLEVELS; ++l) {
        const float res = kRes[l];
        const float fx = px * res, fy = py * res, fz = pz * res;
        const float ix = floorf(fx), iy = floorf(fy), iz = floorf(fz);
        const float wx = fx - ix, wy = fy - iy, wz = fz - iz;
        const unsigned ux = (unsigned)(int)ix;
        const unsigned uy = (unsigned)(int)iy;
        const unsigned uz = (unsigned)(int)iz;

        // --- shared hash subexpressions: 2 muls + 3 adds + 4 xors, then xor+and/corner
        const unsigned hx0 = ux, hx1 = ux + 1u;
        const unsigned hy0 = uy * P2, hy1 = hy0 + P2;
        const unsigned hz0 = uz * P3, hz1 = hz0 + P3;
        const unsigned hyz00 = hy0 ^ hz0;   // y=0, z=0
        const unsigned hyz10 = hy1 ^ hz0;   // y=1, z=0
        const unsigned hyz01 = hy0 ^ hz1;   // y=0, z=1
        const unsigned hyz11 = hy1 ^ hz1;   // y=1, z=1
        const unsigned h0 = (hx0 ^ hyz00) & (TSIZE - 1u);
        const unsigned h1 = (hx1 ^ hyz00) & (TSIZE - 1u);
        const unsigned h2 = (hx0 ^ hyz10) & (TSIZE - 1u);
        const unsigned h3 = (hx1 ^ hyz10) & (TSIZE - 1u);
        const unsigned h4 = (hx0 ^ hyz01) & (TSIZE - 1u);
        const unsigned h5 = (hx1 ^ hyz01) & (TSIZE - 1u);
        const unsigned h6 = (hx0 ^ hyz11) & (TSIZE - 1u);
        const unsigned h7 = (hx1 ^ hyz11) & (TSIZE - 1u);

        const float2* __restrict__ tbl = ((const float2*)tables) + (size_t)l * TSIZE;
        const float2 c0 = tbl[h0], c1 = tbl[h1], c2 = tbl[h2], c3 = tbl[h3];
        const float2 c4 = tbl[h4], c5 = tbl[h5], c6 = tbl[h6], c7 = tbl[h7];

        // --- factored trilinear weights: 3 subs + 12 muls
        const float wx0 = 1.f - wx, wy0 = 1.f - wy, wz0 = 1.f - wz;
        const float wxy00 = wx0 * wy0, wxy10 = wx * wy0;
        const float wxy01 = wx0 * wy,  wxy11 = wx * wy;
        const float w0 = wxy00 * wz0, w1 = wxy10 * wz0;
        const float w2c = wxy01 * wz0, w3 = wxy11 * wz0;
        const float w4 = wxy00 * wz,  w5 = wxy10 * wz;
        const float w6 = wxy01 * wz,  w7 = wxy11 * wz;

        float f0, f1;
        f0 = w0 * c0.x;              f1 = w0 * c0.y;
        f0 = fmaf(w1, c1.x, f0);     f1 = fmaf(w1, c1.y, f1);
        f0 = fmaf(w2c, c2.x, f0);    f1 = fmaf(w2c, c2.y, f1);
        f0 = fmaf(w3, c3.x, f0);     f1 = fmaf(w3, c3.y, f1);
        f0 = fmaf(w4, c4.x, f0);     f1 = fmaf(w4, c4.y, f1);
        f0 = fmaf(w5, c5.x, f0);     f1 = fmaf(w5, c5.y, f1);
        f0 = fmaf(w6, c6.x, f0);     f1 = fmaf(w6, c6.y, f1);
        f0 = fmaf(w7, c7.x, f0);     f1 = fmaf(w7, c7.y, f1);

        union { _Float16 h[2]; unsigned u; } pk;
        pk.h[0] = (_Float16)f0;
        pk.h[1] = (_Float16)f1;
        encbits[l] = pk.u;
    }
    // write the full 64-byte row with 4x ds_store_b128
    {
        uint4* row = (uint4*)&sEnc[wave][lane * 32];
        #pragma unroll
        for (int q = 0; q < 4; ++q) {
            row[q] = make_uint4(encbits[4 * q + 0], encbits[4 * q + 1],
                                encbits[4 * q + 2], encbits[4 * q + 3]);
        }
    }

    __syncthreads();

    // ---- MLP: enc[32,32] @ W1[32,64] via v_wmma_f32_16x16x32_f16 (SRC2 = inline 0),
    //      bias+relu fused in epilogue, then @W2 dot with butterfly reduce ----
    const float b2v = b2[0];
    const int   nl  = lane & 15;               // N column / M row within tile
    const int   kb  = (lane < 16) ? 0 : 8;     // K base per ISA 16-bit A/B layout
    const _Float16* __restrict__ encT = &sEnc[wave][0];

    // B fragments: two ds_load_b128 per tile, contiguous per lane
    v16h bfrag[4];
    #pragma unroll
    for (int nt = 0; nt < 4; ++nt) {
        const v8h* bp = (const v8h*)&sW1frag[nt][lane][0];
        const v8h blo = bp[0];
        const v8h bhi = bp[1];
        bfrag[nt] = __builtin_shufflevector(blo, bhi, 0, 1, 2, 3, 4, 5, 6, 7,
                                            8, 9, 10, 11, 12, 13, 14, 15);
    }

    #pragma unroll
    for (int mt = 0; mt < 2; ++mt) {
        const int mrow = mt * 16 + nl;
        const _Float16* arow = encT + mrow * 32 + kb;
        const v8h alo = *(const v8h*)(arow);        // K = kb .. kb+7
        const v8h ahi = *(const v8h*)(arow + 16);   // K = kb+16 .. kb+23
        const v16h a = __builtin_shufflevector(alo, ahi, 0, 1, 2, 3, 4, 5, 6, 7,
                                               8, 9, 10, 11, 12, 13, 14, 15);

        float partial[8];
        #pragma unroll
        for (int r = 0; r < 8; ++r) partial[r] = 0.f;

        #pragma unroll
        for (int nt = 0; nt < 4; ++nt) {
            const int n = nt * 16 + nl;
            v8f c = {};   // SRC2 encoded as inline constant 0
            c = __builtin_amdgcn_wmma_f32_16x16x32_f16(false, a, false, bfrag[nt],
                                                       (short)0, c, false, false);
            const float binit = sB1[n];
            const float w2n   = sW2[n];
            #pragma unroll
            for (int r = 0; r < 8; ++r) {
                // relu(h + b1) in one v_med3_f32: median(x, 0, +inf) == clamp to [0, inf)
                const float hrelu = __builtin_amdgcn_fmed3f(c[r] + binit, 0.0f,
                                                            __builtin_inff());
                partial[r] = fmaf(hrelu, w2n, partial[r]);
            }
        }

        // reduce the 64-wide N dimension: butterfly within each 16-lane half
        #pragma unroll
        for (int r = 0; r < 8; ++r) {
            float v = partial[r];
            v += __shfl_xor(v, 1);
            v += __shfl_xor(v, 2);
            v += __shfl_xor(v, 4);
            v += __shfl_xor(v, 8);
            partial[r] = v;
        }
        if (nl == 0) {
            // lane 0 holds rows M = mt*16 + 0..7, lane 16 holds M = mt*16 + 8..15
            const int mbase = blockIdx.x * NPB + wave * 32 + mt * 16 + ((lane >> 4) << 3);
            #pragma unroll
            for (int r = 0; r < 8; ++r) {
                const int gi = mbase + r;
                if (gi < npts) out[gi] = partial[r] + b2v;
            }
        }
    }
}

extern "C" void kernel_launch(void* const* d_in, const int* in_sizes, int n_in,
                              void* d_out, int out_size, void* d_ws, size_t ws_size,
                              hipStream_t stream) {
    const float* x      = (const float*)d_in[0];
    const float* tables = (const float*)d_in[1];
    const float* W1     = (const float*)d_in[2];
    const float* b1     = (const float*)d_in[3];
    const float* W2     = (const float*)d_in[4];
    const float* b2     = (const float*)d_in[5];
    float* out = (float*)d_out;

    const int npts = in_sizes[0] / 3;
    const int grid = (npts + NPB - 1) / NPB;
    HashGridMLP_76192719832104_kernel<<<grid, 256, 0, stream>>>(
        x, tables, W1, b1, W2, b2, out, npts);
}
